// LayerStack_5609227288789
// MI455X (gfx1250) — matrix operand order repaired and checked by reading, not compiled, not run on hardware
//
#include <hip/hip_runtime.h>
#include <hip/hip_bf16.h>
#include <cstdint>

// ---------------- problem constants (match reference) ----------------
#define NB      4          // batch
#define TT      16384      // time steps
#define CHN     128        // channels
#define TWO_CH  256        // 2*channels
#define NLAY    10         // dilation layers
#define TN      64         // time tile per block
#define NTILE   (TN / 16)  // N-tiles per block (4)
#define GUARD   1024       // zero rows left of t=0 (max causal reach = 2*512)
#define TROW    (TT + GUARD)

typedef _Float16 v16h  __attribute__((ext_vector_type(16)));
typedef float    v8f   __attribute__((ext_vector_type(8)));
typedef uint32_t u32x4 __attribute__((ext_vector_type(4)));
typedef uint32_t u32x8 __attribute__((ext_vector_type(8)));

__device__ __forceinline__ v8f v8f_zero() {
    v8f z;
    #pragma unroll
    for (int i = 0; i < 8; ++i) z[i] = 0.0f;
    return z;
}

// A-fragment K index mapping for 16-bit 16x32 A: contiguous k = 16*(lane/16) + h (matches B fills)
__device__ __host__ __forceinline__ int a_kmap(int lane, int h) {
    return ((lane >> 4) << 4) + h;
}

// ------------------------------------------------------------------
// Pre-swizzle conv weights into A-fragment layout:
// WfA[li][tap][mt(16)][kc(4)][lane(32)][h(16)]  (f16)
// ------------------------------------------------------------------
__global__ __launch_bounds__(256) void prep_convw(const float* __restrict__ conv_w,
                                                  _Float16* __restrict__ WfA) {
    int id = blockIdx.x * 256 + threadIdx.x;       // 983040 total
    int h    = id & 15;
    int r    = id >> 4;
    int lane = r & 31;  r >>= 5;
    int kc   = r & 3;   r >>= 2;
    int mt   = r & 15;  r >>= 4;
    int k    = r % 3;
    int li   = r / 3;
    int M = mt * 16 + (lane & 15);
    int K = kc * 32 + a_kmap(lane, h);
    WfA[id] = (_Float16)conv_w[(((size_t)li * TWO_CH + M) * CHN + K) * 3 + k];
}

// Wf1[li][mt(8)][kc(4)][lane(32)][h(16)] <- w1x1 [L, CH, CH, 1]
__global__ __launch_bounds__(256) void prep_w1x1(const float* __restrict__ w1x1,
                                                 _Float16* __restrict__ Wf1) {
    int id = blockIdx.x * 256 + threadIdx.x;       // 163840 total
    int h    = id & 15;
    int r    = id >> 4;
    int lane = r & 31;  r >>= 5;
    int kc   = r & 3;   r >>= 2;
    int mt   = r & 7;   r >>= 3;
    int li   = r;
    int M = mt * 16 + (lane & 15);
    int K = kc * 32 + a_kmap(lane, h);
    Wf1[id] = (_Float16)w1x1[((size_t)li * CHN + M) * CHN + K];
}

// zero the causal guard rows of both f16 shadow buffers (every call)
__global__ __launch_bounds__(256) void zero_guards(_Float16* __restrict__ h16a,
                                                   _Float16* __restrict__ h16b) {
    size_t i  = (size_t)blockIdx.x * 256 + threadIdx.x;     // 131072 total
    int buf   = (int)(i >> 16);
    size_t j  = i & 65535;
    int b     = (int)(j / (GUARD * CHN / 8));
    size_t r  = j % (GUARD * CHN / 8);
    _Float16* p = (buf ? h16b : h16a) + (size_t)b * TROW * CHN + r * 8;
    *(float4*)p = make_float4(0.f, 0.f, 0.f, 0.f);
}

// h[b,ch,t] = w_re[ch] * x[b,t]; write f32 (channel-major) + f16 (time-major)
__global__ __launch_bounds__(256) void rechannel(const float* __restrict__ x,
                                                 const float* __restrict__ w_re,
                                                 float* __restrict__ h32,
                                                 _Float16* __restrict__ h16) {
    size_t idx = (size_t)blockIdx.x * 256 + threadIdx.x;    // B*CH*T
    int t  = (int)(idx % TT);
    int ch = (int)((idx / TT) % CHN);
    int b  = (int)(idx / ((size_t)TT * CHN));
    float v = w_re[ch] * x[(size_t)b * TT + t];
    h32[idx] = v;
    h16[((size_t)b * TROW + GUARD + t) * CHN + ch] = (_Float16)v;
}

__global__ __launch_bounds__(256) void out_init(const float* __restrict__ b_head,
                                                float* __restrict__ outp) {
    int i = blockIdx.x * 256 + threadIdx.x;                 // B*T
    outp[i] = b_head[0];
}

// ------------------------------------------------------------------
// One WaveNet layer, 64-timestep tile, 8 waves.
// Tap tiles (3 x 16KB contiguous) staged by the Tensor Data Mover:
// wave 0 issues 3 TENSOR_LOAD_TO_LDS descriptors, waits TENSORcnt==0,
// then arrives at the workgroup barrier -> tiles visible to all waves.
// ------------------------------------------------------------------
__global__ __launch_bounds__(256) void layer_kernel(
    const _Float16* __restrict__ h16_in, const float* __restrict__ h32_in,
    _Float16* __restrict__ h16_out, float* __restrict__ h32_out,
    const _Float16* __restrict__ WfA, const _Float16* __restrict__ Wf1,
    const float* __restrict__ conv_b, const float* __restrict__ mix_w,
    const float* __restrict__ b1x1, const float* __restrict__ w_head,
    const float* __restrict__ csig, float* __restrict__ outacc,
    int li, int d)
{
    __shared__ __align__(32) char poolc[3 * TN * CHN * 2];   // 48KB
    _Float16* taps = (_Float16*)poolc;                       // [3][TN][CHN] f16
    float*    z2s  = (float*)poolc;                          // [TN][CHN] f32 (32KB)
    _Float16* post = (_Float16*)(poolc + 32768);             // [TN][CHN] f16 (16KB)
    __shared__ float bias_s[TWO_CH], mix_s[TWO_CH];
    __shared__ float b1_s[CHN], wh_s[CHN], c_s[TN];

    const int tid  = threadIdx.x;
    const int lane = tid & 31;
    const int wv   = tid >> 5;
    const int tilesPerB = TT / TN;
    const int bb = blockIdx.x / tilesPerB;
    const int t0 = (blockIdx.x % tilesPerB) * TN;

    // prefetch this wave's conv A-fragments while staging runs
    __builtin_prefetch(WfA + ((((size_t)li * 3) * 16 + wv * 2) * 4) * 32 * 16, 0, 0);

    // ---- TDM staging: one descriptor per tap tile (contiguous 16KB each).
    if (wv == 0) {
        #pragma unroll
        for (int k = 0; k < 3; ++k) {
            // source row t0-(2-k)*d (guard keeps it in-buffer); 2048 x 8B elements
            const _Float16* gsrc = h16_in +
                ((size_t)bb * TROW + (size_t)(t0 - (2 - k) * d + GUARD)) * CHN;
            uint64_t ga   = (uint64_t)(uintptr_t)gsrc;
            uint32_t ldsa = (uint32_t)(uintptr_t)(taps + k * TN * CHN);

            u32x4 g0;
            g0[0] = 1u;                                   // count=1, user mode
            g0[1] = ldsa;                                 // lds_addr
            g0[2] = (uint32_t)ga;                         // global_addr[31:0]
            g0[3] = (uint32_t)((ga >> 32) & 0x1FFFFFFu)   // global_addr[56:32]
                    | (2u << 30);                         // type=2 ("image")

            u32x8 g1;
            g1[0] = 3u << 16;          // workgroup_mask=0, data_size=3 (8 bytes)
            g1[1] = 2048u << 16;       // tensor_dim0[15:0] (=2048 low half)
            g1[2] = 0u | (1u << 16);   // tensor_dim0 hi=0; tensor_dim1 lo=1
            g1[3] = 0u | (2048u << 16);// tensor_dim1 hi=0; tile_dim0=2048
            g1[4] = 1u;                // tile_dim1=1, tile_dim2=0 (unused)
            g1[5] = 2048u;             // tensor_dim0_stride low32
            g1[6] = 0u;                // stride hi / tensor_dim1_stride lo
            g1[7] = 0u;                // tensor_dim1_stride hi
            asm volatile("tensor_load_to_lds %0, %1"
                         :: "s"(g0), "s"(g1) : "memory");
        }
    }
    if (tid < TWO_CH) {
        bias_s[tid] = conv_b[li * TWO_CH + tid];
        mix_s[tid]  = mix_w[li * TWO_CH + tid];
    }
    if (tid < CHN) { b1_s[tid] = b1x1[li * CHN + tid]; wh_s[tid] = w_head[tid]; }
    if (tid < TN)  { c_s[tid] = csig[(size_t)bb * TT + t0 + tid]; }
    if (wv == 0) __builtin_amdgcn_s_wait_tensorcnt(0);   // TDM done before barrier
    __syncthreads();

    // ---- GEMM1: wave wv owns M-tiles {2wv, 2wv+1} x N-tiles 0..3
    v8f acc[2][NTILE];
    #pragma unroll
    for (int mi = 0; mi < 2; ++mi)
        #pragma unroll
        for (int nt = 0; nt < NTILE; ++nt) acc[mi][nt] = v8f_zero();

    const int m0 = wv * 2;
    #pragma unroll
    for (int k = 0; k < 3; ++k) {
        #pragma unroll
        for (int kc = 0; kc < 4; ++kc) {
            v16h a0 = *(const v16h*)(WfA +
                (((((size_t)li * 3 + k) * 16 + m0    ) * 4 + kc) * 32 + lane) * 16);
            v16h a1 = *(const v16h*)(WfA +
                (((((size_t)li * 3 + k) * 16 + m0 + 1) * 4 + kc) * 32 + lane) * 16);
            #pragma unroll
            for (int nt = 0; nt < NTILE; ++nt) {
                const v16h bfr = *(const v16h*)(taps +
                    (k * TN + nt * 16 + (lane & 15)) * CHN + kc * 32 + ((lane >> 4) << 4));
                acc[0][nt] = __builtin_amdgcn_wmma_f32_16x16x32_f16(
                    false, a0, false, bfr, (short)0, acc[0][nt], false, false);
                acc[1][nt] = __builtin_amdgcn_wmma_f32_16x16x32_f16(
                    false, a1, false, bfr, (short)0, acc[1][nt], false, false);
            }
        }
    }

    // ---- bias + conditioning mix in registers (C layout: M=v+8*(lane>=16), N=lane&15)
    #pragma unroll
    for (int mi = 0; mi < 2; ++mi)
        #pragma unroll
        for (int nt = 0; nt < NTILE; ++nt)
            #pragma unroll
            for (int v = 0; v < 8; ++v) {
                int M  = (m0 + mi) * 16 + v + ((lane >> 4) << 3);
                int tl = nt * 16 + (lane & 15);
                acc[mi][nt][v] += bias_s[M] + mix_s[M] * c_s[tl];
            }
    __syncthreads();   // all waves done reading taps; pool is reusable

    // ---- sigmoid-half waves publish z2 (M in [128,256)) into pool as f32
    if (wv >= 4) {
        #pragma unroll
        for (int mi = 0; mi < 2; ++mi)
            #pragma unroll
            for (int nt = 0; nt < NTILE; ++nt)
                #pragma unroll
                for (int v = 0; v < 8; ++v) {
                    int ch = (m0 + mi) * 16 + v + ((lane >> 4) << 3) - CHN;
                    int tl = nt * 16 + (lane & 15);
                    z2s[tl * CHN + ch] = acc[mi][nt][v];
                }
    }
    __syncthreads();

    // ---- tanh-half waves: post = tanh(z1) * sigmoid(z2) -> f16 tile
    if (wv < 4) {
        #pragma unroll
        for (int mi = 0; mi < 2; ++mi)
            #pragma unroll
            for (int nt = 0; nt < NTILE; ++nt)
                #pragma unroll
                for (int v = 0; v < 8; ++v) {
                    int ch = (m0 + mi) * 16 + v + ((lane >> 4) << 3);
                    int tl = nt * 16 + (lane & 15);
                    float z2 = z2s[tl * CHN + ch];
                    float p  = tanhf(acc[mi][nt][v]) * (1.0f / (1.0f + __expf(-z2)));
                    post[tl * CHN + ch] = (_Float16)p;
                }
    }
    __syncthreads();

    // ---- GEMM2: r[128 x TN] = W1x1 @ post; wave wv -> M-tile wv, N-tiles 0..3
    v8f acc2[NTILE];
    #pragma unroll
    for (int nt = 0; nt < NTILE; ++nt) acc2[nt] = v8f_zero();
    #pragma unroll
    for (int kc = 0; kc < 4; ++kc) {
        v16h a = *(const v16h*)(Wf1 +
            ((((size_t)li * 8 + wv) * 4 + kc) * 32 + lane) * 16);
        #pragma unroll
        for (int nt = 0; nt < NTILE; ++nt) {
            const v16h bfr = *(const v16h*)(post +
                (nt * 16 + (lane & 15)) * CHN + kc * 32 + ((lane >> 4) << 4));
            acc2[nt] = __builtin_amdgcn_wmma_f32_16x16x32_f16(
                false, a, false, bfr, (short)0, acc2[nt], false, false);
        }
    }

    // residual add + write h (f32 channel-major, f16 time-major shadow)
    #pragma unroll
    for (int nt = 0; nt < NTILE; ++nt)
        #pragma unroll
        for (int v = 0; v < 8; ++v) {
            int ch = wv * 16 + v + ((lane >> 4) << 3);
            int t  = t0 + nt * 16 + (lane & 15);
            size_t gi = ((size_t)bb * CHN + ch) * TT + t;
            float hn = h32_in[gi] + acc2[nt][v] + b1_s[ch];
            h32_out[gi] = hn;
            h16_out[((size_t)bb * TROW + GUARD + t) * CHN + ch] = (_Float16)hn;
        }

    // head contribution: out[b, t] += sum_ch w_head[ch] * post[t][ch]
    if (tid < TN) {
        float s = 0.0f;
        #pragma unroll 8
        for (int ch = 0; ch < CHN; ++ch)
            s += wh_s[ch] * (float)post[tid * CHN + ch];
        outacc[(size_t)bb * TT + t0 + tid] += s;
    }
}

// ------------------------------------------------------------------
extern "C" void kernel_launch(void* const* d_in, const int* in_sizes, int n_in,
                              void* d_out, int out_size, void* d_ws, size_t ws_size,
                              hipStream_t stream) {
    (void)in_sizes; (void)n_in; (void)out_size; (void)ws_size;

    const float* x      = (const float*)d_in[0];
    const float* c      = (const float*)d_in[1];
    // d_in[2] = head_init (zeros) -- folded out
    const float* w_re   = (const float*)d_in[3];
    const float* conv_w = (const float*)d_in[4];
    const float* conv_b = (const float*)d_in[5];
    const float* mix_w  = (const float*)d_in[6];
    const float* w1x1   = (const float*)d_in[7];
    const float* b1x1   = (const float*)d_in[8];
    const float* w_head = (const float*)d_in[9];
    const float* b_head = (const float*)d_in[10];

    float* outp  = (float*)d_out;                 // [B*T] head output
    float* h32_0 = outp + (size_t)NB * TT;        // [B*CH*T] final h lives here

    // workspace carve-up
    char* w = (char*)d_ws;
    const size_t S16 = (size_t)NB * TROW * CHN * sizeof(_Float16);
    _Float16* h16a = (_Float16*)w; w += S16;
    _Float16* h16b = (_Float16*)w; w += S16;
    float*    h32b = (float*)w;    w += (size_t)NB * TT * CHN * sizeof(float);
    _Float16* WfA  = (_Float16*)w; w += (size_t)NLAY * 3 * 16 * 4 * 32 * 16 * 2;
    _Float16* Wf1  = (_Float16*)w;

    prep_convw<<<3840, 256, 0, stream>>>(conv_w, WfA);
    prep_w1x1<<<640, 256, 0, stream>>>(w1x1, Wf1);
    zero_guards<<<512, 256, 0, stream>>>(h16a, h16b);
    rechannel<<<(NB * CHN * TT) / 256, 256, 0, stream>>>(x, w_re, h32_0, h16a);
    out_init<<<(NB * TT) / 256, 256, 0, stream>>>(b_head, outp);

    static const int dil[NLAY] = {1, 2, 4, 8, 16, 32, 64, 128, 256, 512};
    for (int li = 0; li < NLAY; ++li) {
        const _Float16* hi = (li & 1) ? h16b : h16a;
        _Float16*       ho = (li & 1) ? h16a : h16b;
        const float* h32i  = (li & 1) ? h32b : h32_0;
        float*       h32o  = (li & 1) ? h32_0 : h32b;
        layer_kernel<<<NB * (TT / TN), 256, 0, stream>>>(
            hi, h32i, ho, h32o, WfA, Wf1,
            conv_b, mix_w, b1x1, w_head, c, outp, li, dil[li]);
    }
}
